// CrossMoCo_67869073212269
// MI455X (gfx1250) — matrix operand order repaired and checked by reading, not compiled, not run on hardware
//
#include <hip/hip_runtime.h>

// CDNA5 / gfx1250 CrossMoCo supervised-contrastive loss.
// - GEMMs on v_wmma_f32_16x16x32_bf16 with hi/lo bf16 split (3 WMMA / K-step)
//   for near-fp32 cosine similarities.
// - Memory branch: 4-wave blocks share B tiles; fp32 tiles stream through a
//   double-buffered LDS stage via GLOBAL_LOAD_ASYNC_TO_LDS_B128 (ASYNCcnt),
//   then one convert pass produces scaled bf16 hi/lo tiles so the WMMA loop
//   is pure ds_load_b128 + v_wmma (keeps VGPR count < 256).
// - Fully deterministic: per-block partial buffers, no atomics.

typedef __attribute__((ext_vector_type(16))) __bf16 v16bf;
typedef __attribute__((ext_vector_type(8)))  float  v8f;
typedef __attribute__((ext_vector_type(4)))  int    v4i;

#define B_SZ  512
#define M_SZ  65536
#define D_SZ  256
#define INV_T 14.285714285714285f   // 1 / 0.07
#define YBLK  128                   // partial slots over M for the memory GEMM
#define TILES_PER_Y (M_SZ / 16 / YBLK)  // 32 column tiles per block
#define ROWS_PER_BLK 64             // 4 waves x 16 rows

// ---- async global->LDS path (guarded; falls back to sync copies) ---------
#if defined(__has_builtin)
#  if __has_builtin(__builtin_amdgcn_global_load_async_to_lds_b128) && \
      __has_builtin(__builtin_amdgcn_s_wait_asynccnt)
#    define HAVE_ASYNC_LDS 1
#  endif
#endif
#ifndef HAVE_ASYNC_LDS
#  define HAVE_ASYNC_LDS 0
#endif

#if HAVE_ASYNC_LDS
// signature (from toolchain diagnostics): (AS1 v4i*, AS3 v4i*, imm off, imm cpol)
__device__ __forceinline__ void async_b128(const void* g, void* l) {
  __builtin_amdgcn_global_load_async_to_lds_b128(
      (__attribute__((address_space(1))) v4i*)(void*)g,
      (__attribute__((address_space(3))) v4i*)l, 0, 0);
}
#endif

// ---- bf16 helpers: round-to-nearest-even split  x ~= hi + lo -------------
__device__ __forceinline__ unsigned short f2bf(float x) {
  unsigned int u = __float_as_uint(x);
  u += 0x7fffu + ((u >> 16) & 1u);
  return (unsigned short)(u >> 16);
}
__device__ __forceinline__ float bf2f(unsigned short h) {
  return __uint_as_float(((unsigned int)h) << 16);
}

union V16 { v16bf v; unsigned int u[8]; };

// ---- kernel 1: normalize q rows, store bf16 hi/lo ------------------------
__global__ void normq_kernel(const float* __restrict__ q,
                             unsigned short* __restrict__ qhi,
                             unsigned short* __restrict__ qlo) {
  int row  = blockIdx.x * (blockDim.x >> 5) + (threadIdx.x >> 5);
  int lane = threadIdx.x & 31;
  if (row >= B_SZ) return;
  const float* r = q + (size_t)row * D_SZ;
  float ss = 0.f;
  for (int k = lane; k < D_SZ; k += 32) { float x = r[k]; ss += x * x; }
  for (int m = 16; m >= 1; m >>= 1) ss += __shfl_xor(ss, m, 32);
  float inv = 1.0f / fmaxf(sqrtf(ss), 1e-8f);
  for (int k = lane; k < D_SZ; k += 32) {
    float x = r[k] * inv;
    unsigned short h = f2bf(x);
    unsigned short l = f2bf(x - bf2f(h));
    qhi[row * D_SZ + k] = h;
    qlo[row * D_SZ + k] = l;
  }
}

// ---- kernel 2: inverse norms of pro_memory rows --------------------------
__global__ void pnorm_kernel(const float* __restrict__ p,
                             float* __restrict__ pinv) {
  int row  = blockIdx.x * (blockDim.x >> 5) + (threadIdx.x >> 5);
  int lane = threadIdx.x & 31;
  if (row >= M_SZ) return;
  const float* r = p + (size_t)row * D_SZ;
  float ss = 0.f;
  for (int k = lane; k < D_SZ; k += 32) { float x = r[k]; ss += x * x; }
  for (int m = 16; m >= 1; m >>= 1) ss += __shfl_xor(ss, m, 32);
  if (lane == 0) pinv[row] = 1.0f / fmaxf(sqrtf(ss), 1e-8f);
}

// ---- WMMA A-tile gather: CDNA5 ISA §7.12.2 16-bit A 16x32 layout ----
// lane L -> m = L%16, half h = L/16.
//   VGPR v<4 : K = h*8 + 2v, +1 ;  v>=4 : K = 16 + h*8 + 2(v-4), +1
__device__ __forceinline__ void gather_a(const unsigned short* __restrict__ la,
                                         int rowoff, int mn, int h, int koff,
                                         V16& a) {
#pragma unroll
  for (int vi = 0; vi < 8; vi++) {
    int ka = koff + ((vi < 4) ? (h * 8 + 2 * vi) : (16 + h * 8 + 2 * (vi - 4)));
    a.u[vi] = *(const unsigned int*)&la[(rowoff + mn) * D_SZ + ka];
  }
}
// B (32x16 bf16 from column-major rows): lane L -> n = L%16, half h.
//   VGPR v<4 : K = h*8 + 2v, +1 ;  v>=4 : K = 16 + h*8 + 2(v-4), +1
__device__ __forceinline__ void gather_b(const unsigned short* __restrict__ lb,
                                         int mn, int h, int koff, V16& b) {
#pragma unroll
  for (int vi = 0; vi < 8; vi++) {
    int kb = koff + ((vi < 4) ? (h * 8 + 2 * vi) : (16 + h * 8 + 2 * (vi - 4)));
    b.u[vi] = *(const unsigned int*)&lb[mn * D_SZ + kb];
  }
}

// ---- kernel 3: memory branch GEMM + fused exp/target accumulation --------
// grid (B/64, YBLK), 128 threads (4 waves). Waves share the B tile; B fp32
// tiles stream through a double-buffered async LDS stage, then one convert
// pass produces the scaled bf16 hi/lo tile consumed by the WMMA loop.
__global__ __launch_bounds__(128) void mem_gemm_kernel(
    const unsigned short* __restrict__ qhi, const unsigned short* __restrict__ qlo,
    const float* __restrict__ pmem, const float* __restrict__ pinv,
    const int* __restrict__ labels, const int* __restrict__ plabels,
    float* __restrict__ memSp, float* __restrict__ memPp, float* __restrict__ memCp) {
  __shared__ unsigned short lah[ROWS_PER_BLK * D_SZ];  // 32 KB
  __shared__ unsigned short lal[ROWS_PER_BLK * D_SZ];  // 32 KB
  __shared__ float stage[2][16 * D_SZ];                // 2 x 16 KB (fp32 async)
  __shared__ unsigned short lbh[16 * D_SZ];            // 8 KB (scaled bf16 hi)
  __shared__ unsigned short lbl[16 * D_SZ];            // 8 KB (scaled bf16 lo)
  const int tid  = threadIdx.x;
  const int wid  = tid >> 5;
  const int lane = tid & 31;
  const int h = lane >> 4, mn = lane & 15;
  const int rowblk = blockIdx.x * ROWS_PER_BLK;
  const int rowoff = wid * 16;              // this wave's A sub-tile

  // ---- prologue: bring in A tiles (bf16 hi/lo) and first B stage --------
  {
    const int jbase0 = (blockIdx.y * TILES_PER_Y) * 16;
#if HAVE_ASYNC_LDS
    const unsigned short* gah = qhi + (size_t)rowblk * D_SZ;
    const unsigned short* gal = qlo + (size_t)rowblk * D_SZ;
    for (int i = tid; i < 2048; i += 128) {   // 32 KB per array, 16 B chunks
      async_b128(gah + i * 8, lah + i * 8);
      async_b128(gal + i * 8, lal + i * 8);
    }
    const float* gsrc = pmem + (size_t)jbase0 * D_SZ;
    for (int i = tid; i < 1024; i += 128)     // 16 KB tile
      async_b128(gsrc + i * 4, &stage[0][i * 4]);
#else
    const uint4* sh = (const uint4*)(qhi + (size_t)rowblk * D_SZ);
    const uint4* sl = (const uint4*)(qlo + (size_t)rowblk * D_SZ);
    uint4* dh = (uint4*)lah; uint4* dl = (uint4*)lal;
    for (int i = tid; i < 2048; i += 128) { dh[i] = sh[i]; dl[i] = sl[i]; }
    const uint4* src = (const uint4*)(pmem + (size_t)jbase0 * D_SZ);
    uint4* dst = (uint4*)&stage[0][0];
    for (int i = tid; i < 1024; i += 128) dst[i] = src[i];
#endif
  }

  int rlab[8];
#pragma unroll
  for (int vi = 0; vi < 8; vi++) rlab[vi] = labels[rowblk + rowoff + vi + 8 * h];

  float sS[8], sP[8], sC[8];
#pragma unroll
  for (int vi = 0; vi < 8; vi++) { sS[vi] = 0.f; sP[vi] = 0.f; sC[vi] = 0.f; }

  int cur = 0;
  for (int t = 0; t < TILES_PER_Y; t++) {
    const int jbase = (blockIdx.y * TILES_PER_Y + t) * 16;
#if HAVE_ASYNC_LDS
    __builtin_amdgcn_s_wait_asynccnt(0);  // stage[cur] (and A on t==0) landed
    __syncthreads();                      // all waves' async portions done;
                                          // also: everyone finished tile t-1,
                                          // so lbh/lbl may be overwritten.
    if (t + 1 < TILES_PER_Y) {            // prefetch next tile into stage[1-cur]
      const float* gsrc = pmem + (size_t)(jbase + 16) * D_SZ;
      float* dst = &stage[cur ^ 1][0];
      for (int i = tid; i < 1024; i += 128)
        async_b128(gsrc + i * 4, dst + i * 4);
    }
#else
    __syncthreads();
    if (t > 0) {
      const uint4* src = (const uint4*)(pmem + (size_t)jbase * D_SZ);
      uint4* dst = (uint4*)&stage[cur][0];
      for (int i = tid; i < 1024; i += 128) dst[i] = src[i];
    }
    __syncthreads();
#endif

    {  // convert pass: stage[cur] fp32 -> scaled bf16 hi/lo tile
      const float4* src = (const float4*)&stage[cur][0];
      for (int i = tid; i < 1024; i += 128) {
        float4 x = src[i];
        int e = i * 4, r = e >> 8, c = e & 255;
        float s  = pinv[jbase + r];
        float f0 = x.x * s, f1 = x.y * s, f2 = x.z * s, f3 = x.w * s;
        unsigned short h0 = f2bf(f0), h1 = f2bf(f1), h2 = f2bf(f2), h3 = f2bf(f3);
        unsigned short l0 = f2bf(f0 - bf2f(h0)), l1 = f2bf(f1 - bf2f(h1));
        unsigned short l2 = f2bf(f2 - bf2f(h2)), l3 = f2bf(f3 - bf2f(h3));
        *(unsigned int*)&lbh[r * D_SZ + c]     = (unsigned int)h0 | ((unsigned int)h1 << 16);
        *(unsigned int*)&lbh[r * D_SZ + c + 2] = (unsigned int)h2 | ((unsigned int)h3 << 16);
        *(unsigned int*)&lbl[r * D_SZ + c]     = (unsigned int)l0 | ((unsigned int)l1 << 16);
        *(unsigned int*)&lbl[r * D_SZ + c + 2] = (unsigned int)l2 | ((unsigned int)l3 << 16);
      }
    }
    __syncthreads();

    v8f acc = {};
#pragma unroll
    for (int kk = 0; kk < 8; kk++) {
      V16 ah, al, bh, bl;
      gather_a(lah, rowoff, mn, h, kk * 32, ah);
      gather_a(lal, rowoff, mn, h, kk * 32, al);
      gather_b(lbh, mn, h, kk * 32, bh);
      gather_b(lbl, mn, h, kk * 32, bl);
      // hi*hi + hi*lo + lo*hi  ~  fp32 dot product
      acc = __builtin_amdgcn_wmma_f32_16x16x32_bf16(false, ah.v, false, bh.v, (short)0, acc, false, false);
      acc = __builtin_amdgcn_wmma_f32_16x16x32_bf16(false, ah.v, false, bl.v, (short)0, acc, false, false);
      acc = __builtin_amdgcn_wmma_f32_16x16x32_bf16(false, al.v, false, bh.v, (short)0, acc, false, false);
    }

    // C layout: lane holds column n = mn, rows m = vi + 8*h
    const int plab = plabels[jbase + mn];
#pragma unroll
    for (int vi = 0; vi < 8; vi++) {
      float logit = acc[vi] * INV_T;      // reference: memory logits unshifted
      sS[vi] += __expf(logit);
      if (rlab[vi] == plab) { sP[vi] += logit; sC[vi] += 1.0f; }
    }
    cur ^= 1;
  }

  // reduce across the 16 lanes of each half (xor masks stay within the half)
#pragma unroll
  for (int vi = 0; vi < 8; vi++) {
    for (int m = 8; m >= 1; m >>= 1) {
      sS[vi] += __shfl_xor(sS[vi], m, 32);
      sP[vi] += __shfl_xor(sP[vi], m, 32);
      sC[vi] += __shfl_xor(sC[vi], m, 32);
    }
  }
  if (mn == 0) {
#pragma unroll
    for (int vi = 0; vi < 8; vi++) {
      int row  = rowblk + rowoff + vi + 8 * h;
      int slot = blockIdx.y * B_SZ + row;  // deterministic partials, no atomics
      memSp[slot] = sS[vi];
      memPp[slot] = sP[vi];
      memCp[slot] = sC[vi];
    }
  }
}

// ---- kernel 4: in-batch (src) similarity matrix, WMMA --------------------
__global__ __launch_bounds__(32) void src_gemm_kernel(
    const unsigned short* __restrict__ qhi, const unsigned short* __restrict__ qlo,
    float* __restrict__ srcmat) {
  __shared__ unsigned short lah[16 * D_SZ], lal[16 * D_SZ];
  __shared__ unsigned short lbh[16 * D_SZ], lbl[16 * D_SZ];
  const int lane = threadIdx.x;
  const int h = lane >> 4, mn = lane & 15;
  const int rowbase = blockIdx.x * 16;
  const int colbase = blockIdx.y * 16;
  {
    const uint4* sah = (const uint4*)(qhi + rowbase * D_SZ);
    const uint4* sal = (const uint4*)(qlo + rowbase * D_SZ);
    const uint4* sbh = (const uint4*)(qhi + colbase * D_SZ);
    const uint4* sbl = (const uint4*)(qlo + colbase * D_SZ);
    uint4* dah = (uint4*)lah; uint4* dal = (uint4*)lal;
    uint4* dbh = (uint4*)lbh; uint4* dbl = (uint4*)lbl;
    for (int i = lane; i < 512; i += 32) {
      dah[i] = sah[i]; dal[i] = sal[i]; dbh[i] = sbh[i]; dbl[i] = sbl[i];
    }
  }
  __syncthreads();
  v8f acc = {};
#pragma unroll
  for (int kk = 0; kk < 8; kk++) {
    V16 ah, al, bh, bl;
    gather_a(lah, 0, mn, h, kk * 32, ah);
    gather_a(lal, 0, mn, h, kk * 32, al);
    gather_b(lbh, mn, h, kk * 32, bh);
    gather_b(lbl, mn, h, kk * 32, bl);
    acc = __builtin_amdgcn_wmma_f32_16x16x32_bf16(false, ah.v, false, bh.v, (short)0, acc, false, false);
    acc = __builtin_amdgcn_wmma_f32_16x16x32_bf16(false, ah.v, false, bl.v, (short)0, acc, false, false);
    acc = __builtin_amdgcn_wmma_f32_16x16x32_bf16(false, al.v, false, bh.v, (short)0, acc, false, false);
  }
#pragma unroll
  for (int vi = 0; vi < 8; vi++) {
    int row = rowbase + vi + 8 * h;
    srcmat[(size_t)row * B_SZ + colbase + mn] = acc[vi] * INV_T;
  }
}

// ---- kernel 5: src branch row reductions (max / exp-sum / positives) -----
__global__ void src_reduce_kernel(const float* __restrict__ srcmat,
                                  const int* __restrict__ labels,
                                  float* __restrict__ srcS, float* __restrict__ srcP,
                                  float* __restrict__ srcC) {
  int row  = blockIdx.x * (blockDim.x >> 5) + (threadIdx.x >> 5);
  int lane = threadIdx.x & 31;
  if (row >= B_SZ) return;
  const float* r = srcmat + (size_t)row * B_SZ;
  float mx = -3.0e38f;
  for (int j = lane; j < B_SZ; j += 32) mx = fmaxf(mx, r[j]);
  for (int m = 16; m >= 1; m >>= 1) mx = fmaxf(mx, __shfl_xor(mx, m, 32));
  const int li = labels[row];
  float s = 0.f, p = 0.f, c = 0.f;
  for (int j = lane; j < B_SZ; j += 32) {
    if (j == row) continue;               // diagonal masked
    float lg = r[j] - mx;
    s += __expf(lg);
    if (labels[j] == li) { p += lg; c += 1.0f; }
  }
  for (int m = 16; m >= 1; m >>= 1) {
    s += __shfl_xor(s, m, 32);
    p += __shfl_xor(p, m, 32);
    c += __shfl_xor(c, m, 32);
  }
  if (lane == 0) { srcS[row] = s; srcP[row] = p; srcC[row] = c; }
}

// ---- kernel 6: combine partials -> scalar loss ---------------------------
__global__ __launch_bounds__(512) void finalize_kernel(
    const float* __restrict__ memSp, const float* __restrict__ memPp,
    const float* __restrict__ memCp, const float* __restrict__ srcS,
    const float* __restrict__ srcP, const float* __restrict__ srcC,
    float* __restrict__ out) {
  __shared__ float red[512];
  const int i = threadIdx.x;
  float S = srcS[i], P = srcP[i], C = srcC[i];
  for (int k = 0; k < YBLK; k++) {
    S += memSp[k * B_SZ + i];
    P += memPp[k * B_SZ + i];
    C += memCp[k * B_SZ + i];
  }
  red[i] = P / C - __logf(S);   // mean_log_prob_pos[i]
  __syncthreads();
  for (int s = 256; s >= 1; s >>= 1) {
    if (i < s) red[i] += red[i + s];
    __syncthreads();
  }
  if (i == 0) out[0] = -red[0] / (float)B_SZ;
}

// ---- host side -----------------------------------------------------------
extern "C" void kernel_launch(void* const* d_in, const int* in_sizes, int n_in,
                              void* d_out, int out_size, void* d_ws, size_t ws_size,
                              hipStream_t stream) {
  (void)in_sizes; (void)n_in; (void)out_size; (void)ws_size;
  const float* q       = (const float*)d_in[0];
  const int*   labels  = (const int*)d_in[1];
  const float* pmem    = (const float*)d_in[2];
  const int*   plabels = (const int*)d_in[3];
  float* out = (float*)d_out;

  // workspace layout (~2.51 MB total)
  char* ws = (char*)d_ws;
  unsigned short* qhi  = (unsigned short*)(ws);                       // 256 KB
  unsigned short* qlo  = (unsigned short*)(ws + (256u << 10));        // 256 KB
  float* pinv          = (float*)(ws + (512u << 10));                 // 256 KB
  float* memSp         = (float*)(ws + (768u  << 10));                // 256 KB
  float* memPp         = (float*)(ws + (1024u << 10));                // 256 KB
  float* memCp         = (float*)(ws + (1280u << 10));                // 256 KB
  float* srcS          = (float*)(ws + (1536u << 10));                // 2 KB
  float* srcP          = (float*)(ws + (1536u << 10) + 2048);         // 2 KB
  float* srcC          = (float*)(ws + (1536u << 10) + 4096);         // 2 KB
  float* srcmat        = (float*)(ws + (1544u << 10));                // 1 MB

  normq_kernel<<<dim3(B_SZ / 8), dim3(256), 0, stream>>>(q, qhi, qlo);
  pnorm_kernel<<<dim3(M_SZ / 8), dim3(256), 0, stream>>>(pmem, pinv);
  mem_gemm_kernel<<<dim3(B_SZ / ROWS_PER_BLK, YBLK), dim3(128), 0, stream>>>(
      qhi, qlo, pmem, pinv, labels, plabels, memSp, memPp, memCp);
  src_gemm_kernel<<<dim3(B_SZ / 16, B_SZ / 16), dim3(32), 0, stream>>>(qhi, qlo, srcmat);
  src_reduce_kernel<<<dim3(B_SZ / 8), dim3(256), 0, stream>>>(srcmat, labels, srcS, srcP, srcC);
  finalize_kernel<<<dim3(1), dim3(512), 0, stream>>>(memSp, memPp, memCp, srcS, srcP, srcC, out);
}
